// PCDAlignment_36455682409117
// MI455X (gfx1250) — matrix operand order repaired and checked by reading, not compiled, not run on hardware
//
#include <hip/hip_runtime.h>
#include <hip/hip_bf16.h>

typedef __attribute__((ext_vector_type(16))) _Float16 v16h;
typedef __attribute__((ext_vector_type(8)))  _Float16 v8h;
typedef __attribute__((ext_vector_type(8)))  float    v8f;

#define BB 4
#define CC 64
#define HH 128
#define WW 128
#define HW (HH*WW)
#define DGRP 8
#define KTAP 9
#define CG 8

#define TN 64      // pixels per block (one row segment; W/TN == 2)
#define KC 32      // K chunk per WMMA step
#define LDA 40     // padded LDS row stride in f16 (80B, 16B aligned, conflict-spread)

__device__ __forceinline__ float lrelu_f(float v) { return v >= 0.f ? v : 0.1f * v; }
__device__ __forceinline__ int clampi(int v, int lo, int hi) { return v < lo ? lo : (v > hi ? hi : v); }

// ---------------------------------------------------------------------------
// Implicit-GEMM 3x3 conv (pad=1), fp32 in/out, f16 WMMA core, LDS double-buffer.
// Block: 256 threads = 8 waves. Tile: 64 outch x 64 px. Virtual concat via x1/x2.
// grid.x = BB*HH*(WW/TN); grid.y = ceil(Cout/64). Requires (C1+C2)*9 % 32 == 0.
// ---------------------------------------------------------------------------
__global__ __launch_bounds__(256)
void conv3x3_igemm_wmma(const float* __restrict__ x1, int C1,
                        const float* __restrict__ x2, int C2,
                        const float* __restrict__ w, const float* __restrict__ bias,
                        float* __restrict__ y, int Cout, int act)
{
    __shared__ __align__(16) _Float16 Ash[2][64 * LDA];
    __shared__ __align__(16) _Float16 Bsh[2][TN * LDA];

    const int tid  = threadIdx.x;
    const int wv   = tid >> 5;
    const int lane = tid & 31;

    const int bidx  = blockIdx.x;
    const int xb    = (bidx & 1) * TN;
    const int yrow  = (bidx >> 1) & (HH - 1);
    const int b     = bidx >> 8;              // /(2*128)
    const int mbase = blockIdx.y * 64;

    const int Cin  = C1 + C2;
    const int Ktot = Cin * 9;                 // multiple of 32 for all call sites

    // staging: thread -> (row srow, 8-wide k slice at skk)
    const int srow = tid & 63;
    const int skk  = (tid >> 6) * 8;

    const int  mg     = mbase + srow;
    const float mscale = (mg < Cout) ? 1.f : 0.f;
    const float* wrow  = w + (size_t)(mg < Cout ? mg : 0) * Ktot;  // row-major (Cout, Cin*9)

    v8f acc0 = {}; v8f acc1 = {};

    const int mt   = wv >> 1;
    const int nt0  = (wv & 1) * 2;
    const int arow = mt * 16 + (lane & 15);
    const int as   = (lane < 16) ? 0 : 8;     // A frag K start (ISA 16-bit A 16x32 layout)
    const int bs   = (lane < 16) ? 0 : 8;     // B frag K start (ISA 16-bit B 32x16 layout)

    float ar[8], br[8];

    auto stageA = [&](int kc0) {
        const float4* p4 = (const float4*)(wrow + kc0 + skk);
        float4 v0 = p4[0], v1 = p4[1];
        ar[0] = v0.x * mscale; ar[1] = v0.y * mscale; ar[2] = v0.z * mscale; ar[3] = v0.w * mscale;
        ar[4] = v1.x * mscale; ar[5] = v1.y * mscale; ar[6] = v1.z * mscale; ar[7] = v1.w * mscale;
    };
    auto stageB = [&](int kc0) {
        #pragma unroll
        for (int j = 0; j < 8; ++j) {
            int k  = kc0 + skk + j;
            int ci = k / 9, p = k - ci * 9;
            int pr = p / 3;
            int yy = yrow + pr - 1;
            int xx = xb + srow + (p - pr * 3) - 1;
            int yc = clampi(yy, 0, HH - 1);
            int xc = clampi(xx, 0, WW - 1);
            float ok = ((yy == yc) & (xx == xc)) ? 1.f : 0.f;
            const float* src = (ci < C1)
                ? x1 + ((size_t)(b * C1 + ci)) * HW
                : x2 + ((size_t)(b * C2 + (ci - C1))) * HW;
            br[j] = src[yc * WW + xc] * ok;     // unconditional clamped load
        }
    };
    auto commit = [&](int buf) {
        v8h ha, hb;
        #pragma unroll
        for (int j = 0; j < 8; ++j) { ha[j] = (_Float16)ar[j]; hb[j] = (_Float16)br[j]; }
        *(v8h*)&Ash[buf][srow * LDA + skk] = ha;
        *(v8h*)&Bsh[buf][srow * LDA + skk] = hb;
    };

    stageA(0); stageB(0); commit(0);
    __syncthreads();

    int cur = 0;
    for (int kc0 = 0; kc0 < Ktot; kc0 += KC) {
        const int  nxt  = kc0 + KC;
        const bool more = nxt < Ktot;
        if (more) { stageA(nxt); stageB(nxt); }   // global loads overlap WMMA below

        v8h alo = *(const v8h*)&Ash[cur][arow * LDA + as];
        v8h ahi = *(const v8h*)&Ash[cur][arow * LDA + as + 16];
        v16h afrag;
        #pragma unroll
        for (int i = 0; i < 8; ++i) { afrag[i] = alo[i]; afrag[8 + i] = ahi[i]; }

        int n0 = nt0 * 16 + (lane & 15);
        v8h b0lo = *(const v8h*)&Bsh[cur][n0 * LDA + bs];
        v8h b0hi = *(const v8h*)&Bsh[cur][n0 * LDA + bs + 16];
        v16h bf0;
        #pragma unroll
        for (int i = 0; i < 8; ++i) { bf0[i] = b0lo[i]; bf0[8 + i] = b0hi[i]; }

        int n1 = n0 + 16;
        v8h b1lo = *(const v8h*)&Bsh[cur][n1 * LDA + bs];
        v8h b1hi = *(const v8h*)&Bsh[cur][n1 * LDA + bs + 16];
        v16h bf1;
        #pragma unroll
        for (int i = 0; i < 8; ++i) { bf1[i] = b1lo[i]; bf1[8 + i] = b1hi[i]; }

        acc0 = __builtin_amdgcn_wmma_f32_16x16x32_f16(false, afrag, false, bf0, (short)0, acc0, false, false);
        acc1 = __builtin_amdgcn_wmma_f32_16x16x32_f16(false, afrag, false, bf1, (short)0, acc1, false, false);

        if (more) commit(cur ^ 1);
        __syncthreads();
        cur ^= 1;
    }

    // ---- store: lanes 0-15 -> M=r, lanes 16-31 -> M=r+8 (ISA C/D layout) ----
    const int half = (lane < 16) ? 0 : 8;
    const int ncol = lane & 15;
    #pragma unroll
    for (int r = 0; r < 8; ++r) {
        int m = mbase + mt * 16 + r + half;
        if (m < Cout) {
            float bval = bias[m];
            int xx0 = xb + nt0 * 16 + ncol;
            float v0 = acc0[r] + bval;
            float v1 = acc1[r] + bval;
            if (act) { v0 = lrelu_f(v0); v1 = lrelu_f(v1); }
            size_t base = (((size_t)b * Cout + m) * HH + yrow) * WW;
            y[base + xx0]      = v0;
            y[base + xx0 + 16] = v1;
        }
    }
}

// ---------------------------------------------------------------------------
// Modulated deformable conv (DCNv2). om = [o1(72) | o2(72) | mask(72)] channels.
// Precompute per-(pixel,group,tap): clamped corner coords + mask*bilinear weights
// with corner validity folded in (invalid corner -> weight 0). Gather loop is
// then 4 unconditional loads + FMAs. GEMM: out[64] = W[64x576] @ val[576x64px].
// ---------------------------------------------------------------------------
struct __align__(4) PreT {
    unsigned char y0c, x0c, y1c, x1c;         // clamped corner coords
    _Float16 w00, w01, w10, w11;              // mask * bilinear * validity
};

__global__ __launch_bounds__(256)
void dcn_igemm_wmma(const float* __restrict__ x, const float* __restrict__ om,
                    const float* __restrict__ w, const float* __restrict__ bias,
                    float* __restrict__ y, int act)
{
    __shared__ PreT pre[TN * DGRP * KTAP];    // 64*72 entries, 55 KB
    __shared__ __align__(16) _Float16 Ash[2][64 * LDA];
    __shared__ __align__(16) _Float16 Bsh[2][TN * LDA];

    const int tid  = threadIdx.x;
    const int wv   = tid >> 5;
    const int lane = tid & 31;

    const int bidx = blockIdx.x;
    const int xb   = (bidx & 1) * TN;
    const int yrow = (bidx >> 1) & (HH - 1);
    const int b    = bidx >> 8;

    // ---- precompute corners + validity-folded mask-scaled bilinear weights ----
    const float* omb = om + (size_t)b * 216 * HW;
    for (int e = tid; e < TN * DGRP * KTAP; e += 256) {
        int n  = e / (DGRP * KTAP);
        int r  = e - n * (DGRP * KTAP);
        int g  = r / KTAP;
        int kk = r - g * KTAP;
        size_t sp = (size_t)yrow * WW + (xb + n);
        float dy = omb[(g * 18 + kk * 2 + 0) * HW + sp];
        float dx = omb[(g * 18 + kk * 2 + 1) * HW + sp];
        float mv = omb[(144 + g * 9 + kk) * HW + sp];
        float mask = 1.f / (1.f + __expf(-mv));
        float py = (float)yrow + (float)(kk / 3 - 1) + dy;
        float px = (float)(xb + n) + (float)(kk % 3 - 1) + dx;
        float fy = floorf(py), fx = floorf(px);
        float ty = py - fy, tx = px - fx;
        int iy0 = clampi((int)fy, -1, HH);    // keep relative validity, bound range
        int ix0 = clampi((int)fx, -1, WW);
        int iy1 = iy0 + 1, ix1 = ix0 + 1;
        float y0ok = (iy0 >= 0 && iy0 < HH) ? 1.f : 0.f;
        float y1ok = (iy1 >= 0 && iy1 < HH) ? 1.f : 0.f;
        float x0ok = (ix0 >= 0 && ix0 < WW) ? 1.f : 0.f;
        float x1ok = (ix1 >= 0 && ix1 < WW) ? 1.f : 0.f;
        PreT p;
        p.y0c = (unsigned char)clampi(iy0, 0, HH - 1);
        p.x0c = (unsigned char)clampi(ix0, 0, WW - 1);
        p.y1c = (unsigned char)clampi(iy1, 0, HH - 1);
        p.x1c = (unsigned char)clampi(ix1, 0, WW - 1);
        p.w00 = (_Float16)((1.f - ty) * (1.f - tx) * mask * y0ok * x0ok);
        p.w01 = (_Float16)((1.f - ty) * tx * mask * y0ok * x1ok);
        p.w10 = (_Float16)(ty * (1.f - tx) * mask * y1ok * x0ok);
        p.w11 = (_Float16)(ty * tx * mask * y1ok * x1ok);
        pre[e] = p;
    }
    __syncthreads();

    v8f acc0 = {}; v8f acc1 = {};

    const int mt   = wv >> 1;
    const int nt0  = (wv & 1) * 2;
    const int arow = mt * 16 + (lane & 15);
    const int as   = (lane < 16) ? 0 : 8;
    const int bs   = (lane < 16) ? 0 : 8;

    const int srow = tid & 63;
    const int skk  = (tid >> 6) * 8;

    const float* xbp  = x + (size_t)b * CC * HW;
    const float* wrow = w + (size_t)srow * 576;   // W is (64, 576), k = ci*9+tap

    float ar[8], br[8];

    auto stageA = [&](int kc0) {
        const float4* p4 = (const float4*)(wrow + kc0 + skk);
        float4 v0 = p4[0], v1 = p4[1];
        ar[0]=v0.x; ar[1]=v0.y; ar[2]=v0.z; ar[3]=v0.w;
        ar[4]=v1.x; ar[5]=v1.y; ar[6]=v1.z; ar[7]=v1.w;
    };
    auto stageB = [&](int kc0) {
        #pragma unroll
        for (int j = 0; j < 8; ++j) {
            int k  = kc0 + skk + j;
            int g  = k / 72;
            int rr = k - g * 72;
            int c  = rr / 9;
            int q  = rr - c * 9;
            PreT p = pre[srow * 72 + g * 9 + q];
            const float* src = xbp + (size_t)(g * CG + c) * HW;
            float v00 = src[(int)p.y0c * WW + (int)p.x0c];   // all unconditional
            float v01 = src[(int)p.y0c * WW + (int)p.x1c];
            float v10 = src[(int)p.y1c * WW + (int)p.x0c];
            float v11 = src[(int)p.y1c * WW + (int)p.x1c];
            br[j] = (float)p.w00 * v00 + (float)p.w01 * v01
                  + (float)p.w10 * v10 + (float)p.w11 * v11;
        }
    };
    auto commit = [&](int buf) {
        v8h ha, hb;
        #pragma unroll
        for (int j = 0; j < 8; ++j) { ha[j] = (_Float16)ar[j]; hb[j] = (_Float16)br[j]; }
        *(v8h*)&Ash[buf][srow * LDA + skk] = ha;
        *(v8h*)&Bsh[buf][srow * LDA + skk] = hb;
    };

    stageA(0); stageB(0); commit(0);
    __syncthreads();

    int cur = 0;
    for (int kc0 = 0; kc0 < 576; kc0 += KC) {
        const int  nxt  = kc0 + KC;
        const bool more = nxt < 576;
        if (more) { stageA(nxt); stageB(nxt); }

        v8h alo = *(const v8h*)&Ash[cur][arow * LDA + as];
        v8h ahi = *(const v8h*)&Ash[cur][arow * LDA + as + 16];
        v16h afrag;
        #pragma unroll
        for (int i = 0; i < 8; ++i) { afrag[i] = alo[i]; afrag[8 + i] = ahi[i]; }

        int n0 = nt0 * 16 + (lane & 15);
        v8h b0lo = *(const v8h*)&Bsh[cur][n0 * LDA + bs];
        v8h b0hi = *(const v8h*)&Bsh[cur][n0 * LDA + bs + 16];
        v16h bf0;
        #pragma unroll
        for (int i = 0; i < 8; ++i) { bf0[i] = b0lo[i]; bf0[8 + i] = b0hi[i]; }

        int n1 = n0 + 16;
        v8h b1lo = *(const v8h*)&Bsh[cur][n1 * LDA + bs];
        v8h b1hi = *(const v8h*)&Bsh[cur][n1 * LDA + bs + 16];
        v16h bf1;
        #pragma unroll
        for (int i = 0; i < 8; ++i) { bf1[i] = b1lo[i]; bf1[8 + i] = b1hi[i]; }

        acc0 = __builtin_amdgcn_wmma_f32_16x16x32_f16(false, afrag, false, bf0, (short)0, acc0, false, false);
        acc1 = __builtin_amdgcn_wmma_f32_16x16x32_f16(false, afrag, false, bf1, (short)0, acc1, false, false);

        if (more) commit(cur ^ 1);
        __syncthreads();
        cur ^= 1;
    }

    const int half = (lane < 16) ? 0 : 8;
    const int ncol = lane & 15;
    #pragma unroll
    for (int r = 0; r < 8; ++r) {
        int m = mt * 16 + r + half;
        float bval = bias[m];
        int xx0 = xb + nt0 * 16 + ncol;
        float v0 = acc0[r] + bval;
        float v1 = acc1[r] + bval;
        if (act) { v0 = lrelu_f(v0); v1 = lrelu_f(v1); }
        size_t base = (((size_t)b * CC + m) * HH + yrow) * WW;
        y[base + xx0]      = v0;
        y[base + xx0 + 16] = v1;
    }
}

// ---------------------------------------------------------------------------
extern "C" void kernel_launch(void* const* d_in, const int* in_sizes, int n_in,
                              void* d_out, int out_size, void* d_ws, size_t ws_size,
                              hipStream_t stream)
{
    (void)in_sizes; (void)n_in; (void)out_size; (void)ws_size;
    const float* nbr      = (const float*)d_in[0];
    const float* ref      = (const float*)d_in[1];
    const float* w_off1   = (const float*)d_in[2];
    const float* b_off1   = (const float*)d_in[3];
    const float* w_dcn_om = (const float*)d_in[4];
    const float* b_dcn_om = (const float*)d_in[5];
    const float* w_dcn    = (const float*)d_in[6];
    const float* b_dcn    = (const float*)d_in[7];
    const float* w_feat   = (const float*)d_in[8];
    const float* b_feat   = (const float*)d_in[9];
    const float* w_cas1   = (const float*)d_in[10];
    const float* b_cas1   = (const float*)d_in[11];
    const float* w_cas2   = (const float*)d_in[12];
    const float* b_cas2   = (const float*)d_in[13];
    const float* w_cdcn_om= (const float*)d_in[14];
    const float* b_cdcn_om= (const float*)d_in[15];
    const float* w_cdcn   = (const float*)d_in[16];
    const float* b_cdcn   = (const float*)d_in[17];

    float* ws = (float*)d_ws;
    const size_t F  = (size_t)BB * CC * HW;    // 4.19M floats (16 MB)
    const size_t OM = (size_t)BB * 216 * HW;   // 14.16M floats (56.6 MB)
    float* t_off = ws;                 // offset maps (64 ch)
    float* t_om  = ws + F;             // 216-ch offset/mask conv output
    float* t_f1  = ws + F + OM;        // dcn1 output
    float* t_f2  = t_f1 + F;           // feat after w_feat conv (used twice)
    float* t_tmp = t_f2 + F;           // cascade temp

    dim3 blk(256);
    dim3 grid(BB * HH * (WW / TN), 1);
    dim3 grid4(BB * HH * (WW / TN), 4);   // 216 out channels -> 4 chunks of 64

    // 1. offset = lrelu(conv(concat(nbr, ref), w_off1))
    conv3x3_igemm_wmma<<<grid,  blk, 0, stream>>>(nbr, CC, ref, CC, w_off1, b_off1, t_off, CC, 1);
    // 2. om = conv(offset, w_dcn_om)
    conv3x3_igemm_wmma<<<grid4, blk, 0, stream>>>(t_off, CC, nullptr, 0, w_dcn_om, b_dcn_om, t_om, 216, 0);
    // 3. feat = dcn(nbr, om, w_dcn)
    dcn_igemm_wmma<<<grid, blk, 0, stream>>>(nbr, t_om, w_dcn, b_dcn, t_f1, 0);
    // 4. feat = lrelu(conv(feat, w_feat))
    conv3x3_igemm_wmma<<<grid,  blk, 0, stream>>>(t_f1, CC, nullptr, 0, w_feat, b_feat, t_f2, CC, 1);
    // 5. tmp = lrelu(conv(concat(feat, ref), w_cas1))
    conv3x3_igemm_wmma<<<grid,  blk, 0, stream>>>(t_f2, CC, ref, CC, w_cas1, b_cas1, t_tmp, CC, 1);
    // 6. offset2 = lrelu(conv(tmp, w_cas2))
    conv3x3_igemm_wmma<<<grid,  blk, 0, stream>>>(t_tmp, CC, nullptr, 0, w_cas2, b_cas2, t_off, CC, 1);
    // 7. om2 = conv(offset2, w_cdcn_om)
    conv3x3_igemm_wmma<<<grid4, blk, 0, stream>>>(t_off, CC, nullptr, 0, w_cdcn_om, b_cdcn_om, t_om, 216, 0);
    // 8. out = lrelu(dcn(feat, om2, w_cdcn))
    dcn_igemm_wmma<<<grid, blk, 0, stream>>>(t_f2, t_om, w_cdcn, b_cdcn, (float*)d_out, 1);
}
